// GraphConvLayer_25305947308619
// MI455X (gfx1250) — matrix operand order repaired and checked by reading, not compiled, not run on hardware
//
#include <hip/hip_runtime.h>
#include <hip/hip_bf16.h>

typedef __attribute__((ext_vector_type(16))) _Float16 v16h_t;
typedef __attribute__((ext_vector_type(8)))  _Float16 v8h_t;
typedef __attribute__((ext_vector_type(8)))  float    v8f_t;
typedef __attribute__((ext_vector_type(4)))  float    v4f_t;

#define D 128         // feature dim
#define LN_EPS 1e-5f

// ---------------------------------------------------------------------------
// Kernel 0: zero workspace accumulators (agg + deg). Must run every launch.
// ---------------------------------------------------------------------------
__global__ void zero_ws(float* __restrict__ agg, float* __restrict__ deg,
                        int n_agg, int n_deg) {
    int i = blockIdx.x * blockDim.x + threadIdx.x;
    int stride = gridDim.x * blockDim.x;
    v4f_t z = {};
    int n4 = n_agg >> 2;
    v4f_t* a4 = (v4f_t*)agg;
    for (int k = i; k < n4; k += stride) a4[k] = z;
    for (int k = i; k < n_deg; k += stride) deg[k] = 0.0f;
}

// ---------------------------------------------------------------------------
// Kernel 1: x_t = x @ Wn + bn   (WMMA f16 inputs, f32 accumulate)
// Writes f32 x_t (residual path) and f16 x_t_h (edge-gather path).
// Wn pre-swizzled into LDS as WMMA B-fragments:
//   frag(kt,nt): lane L holds 16 contiguous halves =
//     B[K = kt*32 + (L<16?0:8) + (j&7) + (j>=8?16:0)][n = nt*16 + L%16]
// ---------------------------------------------------------------------------
__global__ void node_transform(const float* __restrict__ x,
                               const float* __restrict__ Wn,
                               const float* __restrict__ bn,
                               float* __restrict__ x_t,
                               _Float16* __restrict__ x_t_h,
                               int N) {
    __shared__ _Float16 sB[32 * 512];   // 4 K-tiles * 8 N-tiles * 512 halves = 32 KB
    __shared__ float sbn[D];

    for (int idx = threadIdx.x; idx < 32 * 512; idx += blockDim.x) {
        int frag = idx >> 9, rem = idx & 511;
        int L = rem >> 4, j = rem & 15;
        int kt = frag >> 3, nt = frag & 7;
        int K = kt * 32 + ((L < 16) ? 0 : 8) + (j & 7) + ((j & 8) << 1);
        int n = nt * 16 + (L & 15);
        sB[idx] = (_Float16)Wn[K * D + n];
    }
    for (int i = threadIdx.x; i < D; i += blockDim.x) sbn[i] = bn[i];
    __syncthreads();
    const v16h_t* sBv = (const v16h_t*)sB;

    const int lane    = threadIdx.x & 31;
    const int mloc    = lane & 15;
    const int halfSel = (lane < 16) ? 0 : 8;     // A: K-offset sel; C/D: row offset
    const int wavesPerBlock = blockDim.x >> 5;
    const int gwave  = blockIdx.x * wavesPerBlock + (threadIdx.x >> 5);
    const int nwaves = gridDim.x * wavesPerBlock;
    const int numTiles = (N + 15) >> 4;

    for (int tile = gwave; tile < numTiles; tile += nwaves) {
        const bool fullTile = (tile * 16 + 16 <= N);   // wave-uniform
        int mrow = tile * 16 + mloc;
        int mclamp = (mrow < N) ? mrow : 0;
        const float* xr = x + (size_t)mclamp * D;

        // Build 4 A-fragments (16x32 f16 each) covering K=0..127
        v16h_t afr[4];
#pragma unroll
        for (int kt = 0; kt < 4; ++kt) {
            int base = kt * 32 + halfSel;                 // {0..7} or {8..15}
            v8f_t lo = *(const v8f_t*)(xr + base);        // K = base .. base+7
            v8f_t hi = *(const v8f_t*)(xr + base + 16);   // K = base+16 .. base+23
            v16h_t a;
#pragma unroll
            for (int j = 0; j < 8; ++j) {
                a[j]     = (_Float16)lo[j];
                a[j + 8] = (_Float16)hi[j];
            }
            afr[kt] = a;
        }

        // Per-lane store base: row (tile*16 + halfSel), column varies per nt.
        const size_t rowBase = (size_t)(tile * 16 + halfSel) * D;

#pragma unroll
        for (int nt = 0; nt < 8; ++nt) {
            v8f_t acc = {};
#pragma unroll
            for (int kt = 0; kt < 4; ++kt) {
                v16h_t b = sBv[(kt * 8 + nt) * 32 + lane];
                acc = __builtin_amdgcn_wmma_f32_16x16x32_f16(
                    false, afr[kt], false, b, (short)0, acc, false, false);
            }
            int n = nt * 16 + mloc;      // C/D: lane -> column
            float bnv = sbn[n];
            if (fullTile) {
                // One base address; v*D folds into the 24-bit instruction offset.
                float*    pt = x_t   + rowBase + n;
                _Float16* ph = x_t_h + rowBase + n;
#pragma unroll
                for (int v = 0; v < 8; ++v) {   // C/D: VGPR v -> row (v + halfSel)
                    float val = acc[v] + bnv;
                    pt[v * D] = val;
                    ph[v * D] = (_Float16)val;
                }
            } else {
#pragma unroll
                for (int v = 0; v < 8; ++v) {
                    int r = tile * 16 + halfSel + v;
                    if (r < N) {
                        float val = acc[v] + bnv;
                        x_t[(size_t)r * D + n]   = val;
                        x_t_h[(size_t)r * D + n] = (_Float16)val;
                    }
                }
            }
        }
    }
}

// ---------------------------------------------------------------------------
// Kernel 2: fused edge transform + message GEMM + scatter-add.
// Per 16-edge tile per wave:
//   A[16 x 256] = [ gather(x_t_h[col]) | edge_attr @ We + be ]  (f16 fragments)
//   msg = relu(A @ Wm + bm)  via 64x v_wmma_f32_16x16x32_f16
//   atomically add msg rows into agg[row], bump deg[row].
// Wm fragments (64 KB) + We/biases in dynamic LDS (69.6 KB < 320 KB WGP).
// ---------------------------------------------------------------------------
__global__ void edge_message(const float* __restrict__ edge_attr,
                             const float* __restrict__ We,
                             const float* __restrict__ be,
                             const float* __restrict__ Wm,
                             const float* __restrict__ bm,
                             const int*   __restrict__ edge_index,
                             const _Float16* __restrict__ x_t_h,
                             float* __restrict__ agg,
                             float* __restrict__ deg,
                             int N, int E) {
    extern __shared__ char smem[];
    _Float16* sWm = (_Float16*)smem;                 // 64 frags * 512 halves = 65536 B
    float*    sWe = (float*)(smem + 65536);          // 6*128 = 768 floats
    float*    sbe = sWe + 768;                       // 128
    float*    sbm = sbe + 128;                       // 128

    for (int idx = threadIdx.x; idx < 64 * 512; idx += blockDim.x) {
        int frag = idx >> 9, rem = idx & 511;
        int L = rem >> 4, j = rem & 15;
        int kt = frag >> 3, nt = frag & 7;
        int K = kt * 32 + ((L < 16) ? 0 : 8) + (j & 7) + ((j & 8) << 1);
        int n = nt * 16 + (L & 15);
        sWm[idx] = (_Float16)Wm[K * D + n];
    }
    for (int i = threadIdx.x; i < 6 * D; i += blockDim.x) sWe[i] = We[i];
    for (int i = threadIdx.x; i < D; i += blockDim.x) { sbe[i] = be[i]; sbm[i] = bm[i]; }
    __syncthreads();
    const v16h_t* sWmv = (const v16h_t*)sWm;

    const int lane    = threadIdx.x & 31;
    const int mloc    = lane & 15;
    const int halfSel = (lane < 16) ? 0 : 8;
    const int wavesPerBlock = blockDim.x >> 5;
    const int gwave  = blockIdx.x * wavesPerBlock + (threadIdx.x >> 5);
    const int nwaves = gridDim.x * wavesPerBlock;
    const int numTiles = (E + 15) >> 4;

    for (int tile = gwave; tile < numTiles; tile += nwaves) {
        const bool fullTile = (tile * 16 + 16 <= E);   // wave-uniform
        int e  = tile * 16 + mloc;
        int ec = (e < E) ? e : 0;

        // Prefetch next tile's indices/attrs (global_prefetch_b8) to hide the
        // index->gather dependency chain.
        int tnext = tile + nwaves;
        if (tnext < numTiles) {
            int en = tnext * 16 + mloc;
            __builtin_prefetch(edge_index + en, 0, 1);
            __builtin_prefetch(edge_index + (size_t)E + en, 0, 1);
            __builtin_prefetch(edge_attr + (size_t)en * 6, 0, 1);
        }

        int col  = edge_index[(size_t)E + ec];   // source node (gather)
        int rowi = edge_index[ec];               // dest node (scatter)

        // --- A fragments, K=0..127: gather x_t_h[col] in f16 (2x b128 per kt) ---
        v16h_t afr[8];
        const _Float16* xr = x_t_h + (size_t)col * D;
#pragma unroll
        for (int kt = 0; kt < 4; ++kt) {
            int base = kt * 32 + halfSel;
            v8h_t lo = *(const v8h_t*)(xr + base);
            v8h_t hi = *(const v8h_t*)(xr + base + 16);
            v16h_t a;
#pragma unroll
            for (int j = 0; j < 8; ++j) { a[j] = lo[j]; a[j + 8] = hi[j]; }
            afr[kt] = a;
        }

        // --- A fragments, K=128..255: e_t = edge_attr @ We + be, inline ---
        float ea[6];
        const float* eap = edge_attr + (size_t)ec * 6;
#pragma unroll
        for (int q = 0; q < 6; ++q) ea[q] = eap[q];
#pragma unroll
        for (int kt = 4; kt < 8; ++kt) {
            int cbase = (kt - 4) * 32 + halfSel;
            v16h_t a;
#pragma unroll
            for (int j = 0; j < 8; ++j) {
                int c0 = cbase + j, c1 = cbase + 16 + j;
                float v0 = sbe[c0], v1 = sbe[c1];
#pragma unroll
                for (int q = 0; q < 6; ++q) {
                    v0 = fmaf(ea[q], sWe[q * D + c0], v0);
                    v1 = fmaf(ea[q], sWe[q * D + c1], v1);
                }
                a[j]     = (_Float16)v0;
                a[j + 8] = (_Float16)v1;
            }
            afr[kt] = a;
        }

        // Degree: one add per edge (lanes 0..15 own distinct edges)
        if (lane < 16 && (fullTile || e < E))
            __hip_atomic_fetch_add(deg + rowi, 1.0f, __ATOMIC_RELAXED,
                                   __HIP_MEMORY_SCOPE_AGENT);

        // Destination base pointers for this lane's 8 output rows
        // (C/D: row = v + halfSel). Hoisted out of the nt loop.
        float* rdst[8];
#pragma unroll
        for (int v = 0; v < 8; ++v) {
            int rv = __shfl(rowi, halfSel + v, 32);
            rdst[v] = agg + (size_t)rv * D;
        }

        // --- GEMM + bias + relu + atomic scatter ---
#pragma unroll
        for (int nt = 0; nt < 8; ++nt) {
            v8f_t acc = {};
#pragma unroll
            for (int kt = 0; kt < 8; ++kt) {
                v16h_t b = sWmv[(kt * 8 + nt) * 32 + lane];
                acc = __builtin_amdgcn_wmma_f32_16x16x32_f16(
                    false, afr[kt], false, b, (short)0, acc, false, false);
            }
            int n = nt * 16 + mloc;
            float bmv = sbm[n];
            if (fullTile) {
                // fast path: unguarded atomic stream, no exec-mask churn
#pragma unroll
                for (int v = 0; v < 8; ++v) {
                    float val = fmaxf(acc[v] + bmv, 0.0f);
                    __hip_atomic_fetch_add(rdst[v] + n, val,
                                           __ATOMIC_RELAXED, __HIP_MEMORY_SCOPE_AGENT);
                }
            } else {
#pragma unroll
                for (int v = 0; v < 8; ++v) {
                    int ee = tile * 16 + halfSel + v;
                    if (ee < E) {
                        float val = fmaxf(acc[v] + bmv, 0.0f);
                        __hip_atomic_fetch_add(rdst[v] + n, val,
                                               __ATOMIC_RELAXED, __HIP_MEMORY_SCOPE_AGENT);
                    }
                }
            }
        }
    }
}

// ---------------------------------------------------------------------------
// Kernel 3: out = LN(agg/max(deg,1) + x_t) * gamma + beta. One wave per node.
// ---------------------------------------------------------------------------
__global__ void finalize_ln(const float* __restrict__ agg,
                            const float* __restrict__ deg,
                            const float* __restrict__ x_t,
                            const float* __restrict__ gamma,
                            const float* __restrict__ beta,
                            float* __restrict__ out, int N) {
    int lane = threadIdx.x & 31;
    int node = blockIdx.x * (blockDim.x >> 5) + (threadIdx.x >> 5);
    if (node >= N) return;

    const float* ar = agg + (size_t)node * D;
    const float* xr = x_t + (size_t)node * D;
    float inv = 1.0f / fmaxf(deg[node], 1.0f);

    float vals[4], s = 0.0f, s2 = 0.0f;
#pragma unroll
    for (int j = 0; j < 4; ++j) {
        int c = lane + j * 32;
        float v = ar[c] * inv + xr[c];
        vals[j] = v; s += v; s2 += v * v;
    }
#pragma unroll
    for (int m = 16; m >= 1; m >>= 1) {
        s  += __shfl_xor(s,  m, 32);
        s2 += __shfl_xor(s2, m, 32);
    }
    float mu  = s * (1.0f / 128.0f);
    float var = s2 * (1.0f / 128.0f) - mu * mu;
    float rs  = rsqrtf(var + LN_EPS);
#pragma unroll
    for (int j = 0; j < 4; ++j) {
        int c = lane + j * 32;
        out[(size_t)node * D + c] = (vals[j] - mu) * rs * gamma[c] + beta[c];
    }
}

// ---------------------------------------------------------------------------
// Launch. Inputs (setup_inputs order):
// 0:x 1:edge_attr 2:Wn 3:bn 4:We 5:be 6:Wm 7:bm 8:gamma 9:beta 10:edge_index
// ---------------------------------------------------------------------------
extern "C" void kernel_launch(void* const* d_in, const int* in_sizes, int n_in,
                              void* d_out, int out_size, void* d_ws, size_t ws_size,
                              hipStream_t stream) {
    const float* x         = (const float*)d_in[0];
    const float* edge_attr = (const float*)d_in[1];
    const float* Wn        = (const float*)d_in[2];
    const float* bn        = (const float*)d_in[3];
    const float* We        = (const float*)d_in[4];
    const float* be        = (const float*)d_in[5];
    const float* Wm        = (const float*)d_in[6];
    const float* bm        = (const float*)d_in[7];
    const float* gamma     = (const float*)d_in[8];
    const float* beta      = (const float*)d_in[9];
    const int*   edge_index= (const int*)d_in[10];

    const int N = in_sizes[0] / D;       // 50000
    const int E = in_sizes[1] / 6;       // 800000

    // Workspace partition (256-byte aligned)
    auto align256 = [](size_t v) { return (v + 255) & ~(size_t)255; };
    char* ws = (char*)d_ws;
    size_t off = 0;
    float*    x_t   = (float*)(ws + off);    off += align256((size_t)N * D * sizeof(float));
    _Float16* x_t_h = (_Float16*)(ws + off); off += align256((size_t)N * D * sizeof(_Float16));
    float*    agg   = (float*)(ws + off);    off += align256((size_t)N * D * sizeof(float));
    float*    degb  = (float*)(ws + off);    off += align256((size_t)N * sizeof(float));
    (void)ws_size; (void)n_in; (void)out_size;

    zero_ws<<<512, 256, 0, stream>>>(agg, degb, N * D, N);

    // 8 waves/block; ~3125 node tiles
    node_transform<<<416, 256, 0, stream>>>(x, Wn, bn, x_t, x_t_h, N);

    // 50000 edge tiles; 1024 blocks * 8 waves, grid-stride. 69632 B dynamic LDS.
    edge_message<<<1024, 256, 69632, stream>>>(edge_attr, We, be, Wm, bm,
                                               edge_index, x_t_h, agg, degb, N, E);

    finalize_ln<<<(N + 7) / 8, 256, 0, stream>>>(agg, degb, x_t, gamma, beta,
                                                 (float*)d_out, N);
}
